// TPC_16415365005366
// MI455X (gfx1250) — matrix operand order repaired and checked by reading, not compiled, not run on hardware
//
#include <hip/hip_runtime.h>
#include <hip/hip_bf16.h>

// ---- problem constants -----------------------------------------------------
#define Tn     512
#define Bn     32
#define TBP    5
#define NROWS  16224          // B*(T-TBP) = 32*507
#define KPACK  1920           // compressed to_los K: 32 flat +144 skip +1728 conv +16 xp
#define XPSTRIDE (16384*16)   // per-layer x_point buffer (B*T x 16 floats)

typedef __attribute__((ext_vector_type(16))) _Float16 v16h;
typedef __attribute__((ext_vector_type(8)))  _Float16 v8h;
typedef __attribute__((ext_vector_type(8)))  float    v8f;

// ===========================================================================
// prep kernels
// ===========================================================================

// point_ws[i] is (16, ind) row-major; pointwise wants lane-z-contiguous (ind,16)
__global__ void transpose_pw_kernel(const float* __restrict__ pw,
                                    float* __restrict__ pwT, int ind) {
  int idx = blockIdx.x * 256 + threadIdx.x;
  if (idx >= ind * 16) return;
  int k = idx >> 4, z = idx & 15;
  pwT[idx] = pw[(size_t)z * ind + k];
}

// Repack to_los_w (128 x 2112) into compressed f16 B-matrix in WMMA-fragment
// order: wtP[((k>>4)*128 + o)*16 + (k&15)], so a lane's 16 B-halves for one
// 16x16x32 step are two contiguous 16-byte loads.
//   k <   32 : flat segment                     W[o, k]
//   k <  176 : skip channels ch (j==0 slots)    W[o, 32 + ch*13]
//   k < 1904 : conv5 channels m=ch*12+jj        W[o, 32 + ch*13 + 1 + jj]
//   k < 1920 : xp5 channels z (13-way folded)   sum_j W[o, 32 + (144+z)*13 + j]
__global__ void pack_tolos_w_kernel(const float* __restrict__ w,
                                    _Float16* __restrict__ wtP) {
  int idx = blockIdx.x * 256 + threadIdx.x;
  if (idx >= KPACK * 128) return;
  int k = idx >> 7, o = idx & 127;
  const float* wo = w + (size_t)o * 2112;
  float v;
  if (k < 32) {
    v = wo[k];
  } else if (k < 176) {
    int ch = k - 32;
    v = wo[32 + ch * 13];
  } else if (k < 1904) {
    int m = k - 176;
    int ch = m / 12, jj = m % 12;
    v = wo[32 + ch * 13 + 1 + jj];
  } else {
    int z = k - 1904;
    const float* p = wo + 32 + (144 + z) * 13;
    v = 0.f;
    #pragma unroll
    for (int j = 0; j < 13; ++j) v += p[j];
  }
  wtP[((size_t)(k >> 4) * 128 + o) * 16 + (k & 15)] = (_Float16)v;
}

__global__ void bn_prep_kernel(const float* __restrict__ g,  const float* __restrict__ be,
                               const float* __restrict__ mean, const float* __restrict__ var,
                               float* __restrict__ scale, float* __restrict__ shift) {
  int o = threadIdx.x;   // 128 threads
  float sc = g[o] * rsqrtf(var[o] + 1e-5f);
  scale[o] = sc;
  shift[o] = be[o] - mean[o] * sc;
}

// ===========================================================================
// grouped dilated conv (layer i): out (B, G*12, T), block = (b,g), 12-way reuse
// ===========================================================================
__global__ void conv_kernel(int layer, int dil, int Gcur, int Gprev, int coutPrev,
                            const float* __restrict__ x, const float* __restrict__ xpAll,
                            const float* __restrict__ convPrev,
                            const float* __restrict__ cw, const float* __restrict__ cb,
                            float* __restrict__ convOut) {
  int b = blockIdx.x / Gcur;
  int g = blockIdx.x % Gcur;
  int cout = Gcur * 12;

  if (layer == 0) {
    // input channels {2g,2g+1} = interleaved (x1[g], x2[g]); no relu on input
    for (int t = (int)threadIdx.x; t < Tn; t += 256) {
      float acc[12];
      #pragma unroll
      for (int y = 0; y < 12; ++y) acc[y] = cb[g * 12 + y];
      #pragma unroll
      for (int k = 0; k < 4; ++k) {
        int tp = t - (3 - k) * dil;
        float v0 = 0.f, v1 = 0.f;
        if (tp >= 0) {
          v0 = x[((size_t)b * 129 + 1 + g) * Tn + tp];
          v1 = x[((size_t)b * 129 + 65 + g) * Tn + tp];
        }
        #pragma unroll
        for (int y = 0; y < 12; ++y) {
          const float* wo = cw + (size_t)(g * 12 + y) * 8;   // (cout,2,4)
          acc[y] += wo[k] * v0 + wo[4 + k] * v1;
        }
      }
      #pragma unroll
      for (int y = 0; y < 12; ++y)
        convOut[((size_t)b * cout + g * 12 + y) * Tn + t] = acc[y];
    }
  } else if (g < Gprev) {
    // c=0 -> relu(skip_g), c=1..12 -> relu(convPrev[g*12+c-1])
    for (int t = (int)threadIdx.x; t < Tn; t += 256) {
      float acc[12];
      #pragma unroll
      for (int y = 0; y < 12; ++y) acc[y] = cb[g * 12 + y];
      #pragma unroll
      for (int k = 0; k < 4; ++k) {
        int tp = t - (3 - k) * dil;
        float v = 0.f;
        if (tp >= 0) {
          if (g < 64) v = x[((size_t)b * 129 + 1 + g) * Tn + tp];
          else {
            int gz = g - 64;
            v = xpAll[(size_t)(gz >> 4) * XPSTRIDE + ((size_t)(b * Tn + tp)) * 16 + (gz & 15)];
          }
          v = v > 0.f ? v : 0.f;
        }
        #pragma unroll
        for (int y = 0; y < 12; ++y)
          acc[y] += cw[(size_t)(g * 12 + y) * 52 + k] * v;     // (cout,13,4)
        for (int c = 1; c < 13; ++c) {
          float u = 0.f;
          if (tp >= 0) {
            u = convPrev[((size_t)b * coutPrev + g * 12 + (c - 1)) * Tn + tp];
            u = u > 0.f ? u : 0.f;
          }
          #pragma unroll
          for (int y = 0; y < 12; ++y)
            acc[y] += cw[(size_t)(g * 12 + y) * 52 + c * 4 + k] * u;
        }
      }
      #pragma unroll
      for (int y = 0; y < 12; ++y)
        convOut[((size_t)b * cout + g * 12 + y) * Tn + t] = acc[y];
    }
  } else {
    // xpr group: all 13 input rows are the same relu(xp_{layer-1}) value
    int zc = g - Gprev;
    float wsum[12][4];
    #pragma unroll
    for (int y = 0; y < 12; ++y)
      #pragma unroll
      for (int k = 0; k < 4; ++k) {
        float s = 0.f;
        for (int c = 0; c < 13; ++c) s += cw[(size_t)(g * 12 + y) * 52 + c * 4 + k];
        wsum[y][k] = s;
      }
    const float* xpPrev = xpAll + (size_t)(layer - 1) * XPSTRIDE;
    for (int t = (int)threadIdx.x; t < Tn; t += 256) {
      float acc[12];
      #pragma unroll
      for (int y = 0; y < 12; ++y) acc[y] = cb[g * 12 + y];
      #pragma unroll
      for (int k = 0; k < 4; ++k) {
        int tp = t - (3 - k) * dil;
        float v = 0.f;
        if (tp >= 0) {
          v = xpPrev[((size_t)(b * Tn + tp)) * 16 + zc];
          v = v > 0.f ? v : 0.f;
        }
        #pragma unroll
        for (int y = 0; y < 12; ++y) acc[y] += wsum[y][k] * v;
      }
      #pragma unroll
      for (int y = 0; y < 12; ++y)
        convOut[((size_t)b * cout + g * 12 + y) * Tn + t] = acc[y];
    }
  }
}

// ===========================================================================
// pointwise (x_cat @ point_w^T + b): thread = (bt, z); raw (no relu) inputs
// ===========================================================================
__global__ void pointwise_kernel(int layer, int coutPrev,
                                 const float* __restrict__ x, const float* __restrict__ flat,
                                 const float* __restrict__ convPrev,
                                 const float* __restrict__ xpAll,
                                 const float* __restrict__ pwT, const float* __restrict__ pb,
                                 float* __restrict__ xpOut) {
  int tid = threadIdx.x;
  int z = tid & 15;
  int bt = blockIdx.x * 16 + (tid >> 4);
  int b = bt >> 9, t = bt & 511;
  float acc = pb[z];
  int off = 0;
  if (layer > 0) {
    for (int m = 0; m < coutPrev; ++m)
      acc += pwT[(size_t)m * 16 + z] * convPrev[((size_t)b * coutPrev + m) * Tn + t];
    off = coutPrev;
    const float* xpPrev = xpAll + (size_t)(layer - 1) * XPSTRIDE + (size_t)bt * 16;
    #pragma unroll
    for (int q = 0; q < 16; ++q)
      acc += pwT[(size_t)(off + q) * 16 + z] * xpPrev[q];
    off += 16;
  }
  for (int c = 0; c < 129; ++c)
    acc += pwT[(size_t)(off + c) * 16 + z] * x[((size_t)b * 129 + c) * Tn + t];
  off += 129;
  #pragma unroll
  for (int f = 0; f < 32; ++f)
    acc += pwT[(size_t)(off + f) * 16 + z] * flat[b * 32 + f];
  xpOut[(size_t)bt * 16 + z] = acc;
}

// ===========================================================================
// compressed-A gather: k in [0,1920) -> relu'd network state, f32
// ===========================================================================
__device__ __forceinline__ float loadA_tolos(int b, int t, int k,
    const float* __restrict__ x, const float* __restrict__ flat,
    const float* __restrict__ conv5, const float* __restrict__ xpAll) {
  if (k < 32) return flat[b * 32 + k];
  k -= 32;
  if (k < 144) {                       // relu(skip): x1(64) + xp0..xp4(80)
    float v;
    if (k < 64) v = x[((size_t)b * 129 + 1 + k) * Tn + t];
    else {
      int z = k - 64;
      v = xpAll[(size_t)(z >> 4) * XPSTRIDE + ((size_t)(b * Tn + t)) * 16 + (z & 15)];
    }
    return v > 0.f ? v : 0.f;
  }
  k -= 144;
  if (k < 1728) {                      // relu(conv5)
    float v = conv5[((size_t)b * 1728 + k) * Tn + t];
    return v > 0.f ? v : 0.f;
  }
  k -= 1728;                           // relu(xp5), weights pre-folded 13x
  float v = xpAll[(size_t)5 * XPSTRIDE + ((size_t)(b * Tn + t)) * 16 + k];
  return v > 0.f ? v : 0.f;
}

// Materialize A (NROWS x KPACK) as row-major f16 once; thread packs 8 K into
// one 16-byte store so the GEMM can use b128 fragment loads.
__global__ void pack_A_kernel(const float* __restrict__ x, const float* __restrict__ flat,
                              const float* __restrict__ conv5, const float* __restrict__ xpAll,
                              _Float16* __restrict__ Af) {
  int idx = blockIdx.x * 256 + threadIdx.x;     // r*(KPACK/8) + kg
  int r = idx / (KPACK / 8);
  int kg = idx % (KPACK / 8);
  if (r >= NROWS) return;
  int b = r / 507;
  int t = r % 507 + TBP;
  v8h v;
  #pragma unroll
  for (int e = 0; e < 8; ++e)
    v[e] = (_Float16)loadA_tolos(b, t, kg * 8 + e, x, flat, conv5, xpAll);
  *(v8h*)(Af + (size_t)r * KPACK + kg * 8) = v;
}

// ===========================================================================
// to_los GEMM on WMMA: A/B in fragment-native f16, 4 x b128 loads per v_wmma
// ===========================================================================
__global__ void gemm_tolos_kernel(
    const _Float16* __restrict__ Af, const _Float16* __restrict__ wtP,
    const float* __restrict__ bnscale, const float* __restrict__ bnshift,
    const float* __restrict__ tolosb, float* __restrict__ outmat) {
  int wid  = threadIdx.x >> 5;     // 8 waves/block = 8 N-tiles (N = 128)
  int lane = threadIdx.x & 31;
  int mt = blockIdx.x;             // 1014 M-tiles (16224/16)
  int hi = lane >> 4;
  int lo = lane & 15;

  int r = mt * 16 + lo;            // A row for this lane (M = lo)
  int n = wid * 16 + lo;           // B/C column
  const _Float16* arow = Af + (size_t)r * KPACK;

  v8f acc = {};
  for (int kt = 0; kt < KPACK / 32; ++kt) {
    // A frag: lanes 0-15 K=ka..ka+15; lanes 16-31 K=ka+16..ka+31
    const _Float16* ap = arow + kt * 32 + hi * 16;
    v8h a0 = *(const v8h*)(ap);
    v8h a1 = *(const v8h*)(ap + 8);
    // B frag: lane n, lanes 0-15 K=kt*32..+15, lanes 16-31 K=kt*32+16..+31
    const _Float16* bp = wtP + ((size_t)(kt * 2 + hi) * 128 + n) * 16;
    v8h b0 = *(const v8h*)(bp);
    v8h b1 = *(const v8h*)(bp + 8);
    v16h a  = __builtin_shufflevector(a0, a1, 0,1,2,3,4,5,6,7,8,9,10,11,12,13,14,15);
    v16h bf = __builtin_shufflevector(b0, b1, 0,1,2,3,4,5,6,7,8,9,10,11,12,13,14,15);
    acc = __builtin_amdgcn_wmma_f32_16x16x32_f16(false, a, false, bf,
                                                 (short)0, acc, false, false);
  }
  // fused bias + batchnorm + relu; C layout: vgpr e -> row M = e + 8*hi, col = lo
  float sc = bnscale[n], sh = bnshift[n], bias = tolosb[n];
  #pragma unroll
  for (int e = 0; e < 8; ++e) {
    int row = mt * 16 + hi * 8 + e;
    float v = (acc[e] + bias) * sc + sh;
    v = v > 0.f ? v : 0.f;
    outmat[(size_t)row * 128 + n] = v;
  }
}

// ===========================================================================
// heads: los = clip(exp(out @ los_w + b)), mort = sigmoid(out @ mort_w + b)
// ===========================================================================
__global__ void heads_kernel(const float* __restrict__ outmat,
                             const float* __restrict__ losw, const float* __restrict__ losb,
                             const float* __restrict__ mortw, const float* __restrict__ mortb,
                             float* __restrict__ out) {
  int r = blockIdx.x * 256 + threadIdx.x;
  if (r >= NROWS) return;
  const float* row = outmat + (size_t)r * 128;
  float dl = 0.f, dm = 0.f;
  #pragma unroll 8
  for (int j = 0; j < 128; ++j) {
    float v = row[j];
    dl += v * losw[j];
    dm += v * mortw[j];
  }
  float losv = expf(dl + losb[0]);
  losv = fminf(fmaxf(losv, 1.0f / 48.0f), 100.0f);
  float mortv = 1.0f / (1.0f + expf(-(dm + mortb[0])));
  out[r] = losv;
  out[NROWS + r] = mortv;
}

// ===========================================================================
extern "C" void kernel_launch(void* const* d_in, const int* in_sizes, int n_in,
                              void* d_out, int out_size, void* d_ws, size_t ws_size,
                              hipStream_t stream) {
  const float* x    = (const float*)d_in[0];
  const float* flat = (const float*)d_in[1];
  const float* conv_w[6]; const float* conv_b[6];
  const float* point_w[6]; const float* point_b[6];
  for (int i = 0; i < 6; ++i) {
    conv_w[i]  = (const float*)d_in[2 + i];
    conv_b[i]  = (const float*)d_in[8 + i];
    point_w[i] = (const float*)d_in[14 + i];
    point_b[i] = (const float*)d_in[20 + i];
  }
  const float* to_los_w = (const float*)d_in[26];
  const float* to_los_b = (const float*)d_in[27];
  const float* bn_gamma = (const float*)d_in[28];
  const float* bn_beta  = (const float*)d_in[29];
  const float* bn_mean  = (const float*)d_in[30];
  const float* bn_var   = (const float*)d_in[31];
  const float* los_w    = (const float*)d_in[32];
  const float* los_b    = (const float*)d_in[33];
  const float* mort_w   = (const float*)d_in[34];
  const float* mort_b   = (const float*)d_in[35];

  // ---- workspace layout (~305 MB) ----
  char* ws = (char*)d_ws;
  size_t off = 0;
  float* convA = (float*)(ws + off); off += (size_t)Bn * 1728 * Tn * 4;
  float* convB = (float*)(ws + off); off += (size_t)Bn * 1728 * Tn * 4;
  float* xpAll = (float*)(ws + off); off += (size_t)6 * XPSTRIDE * 4;
  float* pwT   = (float*)(ws + off); off += (size_t)6806 * 16 * 4;
  _Float16* wtP = (_Float16*)(ws + off); off += (size_t)KPACK * 128 * 2;
  float* bnsc  = (float*)(ws + off); off += 128 * 4;
  float* bnsh  = (float*)(ws + off); off += 128 * 4;
  float* outmat = (float*)(ws + off); off += (size_t)NROWS * 128 * 4;
  _Float16* Af = (_Float16*)(ws + off); off += (size_t)NROWS * KPACK * 2;
  (void)ws_size; (void)in_sizes; (void)n_in; (void)out_size;

  static const int IND[6]  = {161, 945, 1137, 1329, 1521, 1713};
  static const int GARR[6] = {64, 80, 96, 112, 128, 144};
  static const int DIL[6]  = {1, 3, 6, 9, 12, 15};

  // ---- prep ----
  int cum = 0;
  for (int i = 0; i < 6; ++i) {
    int nelem = IND[i] * 16;
    transpose_pw_kernel<<<(nelem + 255) / 256, 256, 0, stream>>>(
        point_w[i], pwT + (size_t)cum * 16, IND[i]);
    cum += IND[i];
  }
  pack_tolos_w_kernel<<<(KPACK * 128 + 255) / 256, 256, 0, stream>>>(to_los_w, wtP);
  bn_prep_kernel<<<1, 128, 0, stream>>>(bn_gamma, bn_beta, bn_mean, bn_var, bnsc, bnsh);

  // ---- layer pipeline ----
  cum = 0;
  for (int i = 0; i < 6; ++i) {
    int G = GARR[i];
    int Gprev = (i == 0) ? 0 : GARR[i - 1];
    int coutPrev = (i == 0) ? 0 : GARR[i - 1] * 12;
    float* cur  = (i & 1) ? convB : convA;
    float* prev = (i & 1) ? convA : convB;
    conv_kernel<<<Bn * G, 256, 0, stream>>>(i, DIL[i], G, Gprev, coutPrev,
        x, xpAll, prev, conv_w[i], conv_b[i], cur);
    pointwise_kernel<<<16384 / 16, 256, 0, stream>>>(i, coutPrev,
        x, flat, prev, xpAll, pwT + (size_t)cum * 16, point_b[i],
        xpAll + (size_t)i * XPSTRIDE);
    cum += IND[i];
  }

  // ---- pack A, head GEMM (WMMA), heads ----  (layer 5 conv lives in convB)
  pack_A_kernel<<<(NROWS * (KPACK / 8) + 255) / 256, 256, 0, stream>>>(
      x, flat, convB, xpAll, Af);
  gemm_tolos_kernel<<<NROWS / 16, 256, 0, stream>>>(
      Af, wtP, bnsc, bnsh, to_los_b, outmat);
  heads_kernel<<<(NROWS + 255) / 256, 256, 0, stream>>>(
      outmat, los_w, los_b, mort_w, mort_b, (float*)d_out);
}
